// WaveletNet_16157666967725
// MI455X (gfx1250) — compile-verified, matrix operand order checked by reading, not run on hardware
//
#include <hip/hip_runtime.h>
#include <hip/hip_bf16.h>

typedef __attribute__((ext_vector_type(16))) _Float16 v16h;
typedef __attribute__((ext_vector_type(8)))  _Float16 v8h;
typedef __attribute__((ext_vector_type(8)))  float    v8f;

#define EPS_BN 2e-5f

// ---------------------------------------------------------------------------
// CDNA5 WMMA 16-bit A/B striping (ISA 7.12.2): lanes 0-15 hold K {0..7,16..23},
// lanes 16-31 hold K {8..15,24..31}; fragment half h -> K index:
__device__ __host__ __forceinline__ int wmma_kmap(int lane, int h) {
    int k = (h < 8) ? h : (h + 8);
    return (lane & 16) ? (k + 8) : k;
}

// ---------------------------------------------------------------------------
__global__ void zero_stats_kernel(float* p, int n) {
    int i = blockIdx.x * blockDim.x + threadIdx.x;
    if (i < n) p[i] = 0.0f;
}

// ---------------------------------------------------------------------------
// Layer 1: lift_conv (9 scales, k=7, 6->12ch) fused with maxpool2 and relu.
// x: (B,100,6); out: (B,12,9,50) f32. Transpose folded into the loads.
__global__ void lift_pool_relu_kernel(const float* __restrict__ x,
                                      const float* __restrict__ w1,
                                      float* __restrict__ out, long total) {
    long g = (long)blockIdx.x * blockDim.x + threadIdx.x;
    if (g >= total) return;
    int co = (int)(g % 12);
    long r = g / 12;
    int t2 = (int)(r % 50); r /= 50;
    int s  = (int)(r % 9);
    long b = r / 9;
    int dil = 1 << s;
    float inv = 1.0f / (float)dil;
    float acc0 = 0.0f, acc1 = 0.0f;
    #pragma unroll
    for (int tt = 0; tt < 2; ++tt) {
        int t = 2 * t2 + tt;
        float acc = 0.0f;
        #pragma unroll
        for (int j = 0; j < 7; ++j) {
            int tp = t + (j - 3) * dil;
            if (tp >= 0 && tp < 100) {
                const float* xp = x + ((long)b * 100 + tp) * 6;
                #pragma unroll
                for (int ci = 0; ci < 6; ++ci)
                    acc += xp[ci] * w1[(co * 6 + ci) * 7 + j];
            }
        }
        if (tt == 0) acc0 = acc; else acc1 = acc;
    }
    float v = fmaxf(acc0, acc1) * inv;
    out[(((long)b * 12 + co) * 9 + s) * 50 + t2] = fmaxf(v, 0.0f);
}

// ---------------------------------------------------------------------------
// gg_conv fused with maxpool2 + relu, templated on (nh,k) for full tap unroll.
// in: (B,Cin,H,X), w: (Cout,Cin,nh,k), out: (B,Cout,Hout,Xout), dilation 2^i.
template <int NH, int KK>
__global__ void ggconv_pool_relu_kernel(const float* __restrict__ in,
                                        const float* __restrict__ w,
                                        float* __restrict__ out,
                                        int Cin, int H, int X, int Cout,
                                        int Hout, int Xout, long total) {
    long g = (long)blockIdx.x * blockDim.x + threadIdx.x;
    if (g >= total) return;
    int co = (int)(g % Cout);
    long r = g / Cout;
    int t2 = (int)(r % Xout); r /= Xout;
    int i  = (int)(r % Hout);
    long b = r / Hout;
    int dil = 1 << i;
    const int kh = KK >> 1;
    float inv = 1.0f / (float)dil;
    float acc0 = 0.0f, acc1 = 0.0f;
    for (int ci = 0; ci < Cin; ++ci) {
        #pragma unroll
        for (int hh = 0; hh < NH; ++hh) {
            const float* ipc = in + (((long)b * Cin + ci) * H + (i + hh)) * X;
            const float* wp  = w + (((long)co * Cin + ci) * NH + hh) * KK;
            #pragma unroll
            for (int j = 0; j < KK; ++j) {
                int off = (j - kh) * dil;
                float wv = wp[j];
                int t0 = 2 * t2 + off;
                int t1 = t0 + 1;
                if (t0 >= 0 && t0 < X) acc0 += ipc[t0] * wv;
                if (t1 >= 0 && t1 < X) acc1 += ipc[t1] * wv;
            }
        }
    }
    float v = fmaxf(acc0, acc1) * inv;
    long oidx = (((long)b * Cout + co) * Hout + i) * Xout + t2;
    out[oidx] = fmaxf(v, 0.0f);
}

// ---------------------------------------------------------------------------
// Per-channel sum / sumsq for batchnorm, layout (B,C,S).
__global__ void stats_kernel(const float* __restrict__ in,
                             float* __restrict__ sum, float* __restrict__ sumsq,
                             int C, int S, long total) {
    __shared__ float ls[128];
    __shared__ float lq[128];
    if (threadIdx.x < 128) { ls[threadIdx.x] = 0.0f; lq[threadIdx.x] = 0.0f; }
    __syncthreads();
    long stride = (long)gridDim.x * blockDim.x;
    for (long i = (long)blockIdx.x * blockDim.x + threadIdx.x; i < total; i += stride) {
        int c = (int)((i / S) % C);
        float v = in[i];
        atomicAdd(&ls[c], v);
        atomicAdd(&lq[c], v * v);
    }
    __syncthreads();
    if (threadIdx.x < (unsigned)C) {
        atomicAdd(&sum[threadIdx.x],   ls[threadIdx.x]);
        atomicAdd(&sumsq[threadIdx.x], lq[threadIdx.x]);
    }
}

// ---------------------------------------------------------------------------
// BN apply fused with maxpool2 over the last dim: (B,C,H,Xin) -> (B,C,H,Xin/2)
__global__ void bn_pool_kernel(const float* __restrict__ in, float* __restrict__ out,
                               const float* __restrict__ sum, const float* __restrict__ sumsq,
                               const float* __restrict__ gam, const float* __restrict__ bet,
                               int C, int H, int Xin, float invN, long totalOut) {
    long idx = (long)blockIdx.x * blockDim.x + threadIdx.x;
    if (idx >= totalOut) return;
    int Xout = Xin >> 1;
    int t2 = (int)(idx % Xout);
    long r = idx / Xout;
    int hh = (int)(r % H); r /= H;
    int c  = (int)(r % C);
    long b = r / C;
    float m  = sum[c] * invN;
    float vv = sumsq[c] * invN - m * m;
    float sc = gam[c] * rsqrtf(vv + EPS_BN);
    float sh = bet[c] - m * sc;
    long base = (((long)b * C + c) * H + hh) * Xin + 2 * t2;
    float v0 = in[base] * sc + sh;
    float v1 = in[base + 1] * sc + sh;
    out[idx] = fmaxf(v0, v1);
}

// BN apply, elementwise (no pool). Layout (B,C,S).
__global__ void bn_apply_kernel(const float* __restrict__ in, float* __restrict__ out,
                                const float* __restrict__ sum, const float* __restrict__ sumsq,
                                const float* __restrict__ gam, const float* __restrict__ bet,
                                int C, int S, float invN, long total) {
    long idx = (long)blockIdx.x * blockDim.x + threadIdx.x;
    if (idx >= total) return;
    int c = (int)((idx / S) % C);
    float m  = sum[c] * invN;
    float vv = sumsq[c] * invN - m * m;
    float sc = gam[c] * rsqrtf(vv + EPS_BN);
    float sh = bet[c] - m * sc;
    out[idx] = in[idx] * sc + sh;
}

// BN apply for layer 4 + flatten (B,96,3,1)->(B,288) + f16 pack for the GEMM.
__global__ void bn_flatten_f16_kernel(const float* __restrict__ in, _Float16* __restrict__ out,
                                      const float* __restrict__ sum, const float* __restrict__ sumsq,
                                      const float* __restrict__ gam, const float* __restrict__ bet,
                                      float invN, long total) {
    long idx = (long)blockIdx.x * blockDim.x + threadIdx.x;
    if (idx >= total) return;
    int f = (int)(idx % 288);
    long b = idx / 288;
    int c = f / 3, i = f % 3;
    float m  = sum[c] * invN;
    float vv = sumsq[c] * invN - m * m;
    float sc = gam[c] * rsqrtf(vv + EPS_BN);
    float sh = bet[c] - m * sc;
    float v = in[((long)b * 96 + c) * 3 + i] * sc + sh;
    out[idx] = (_Float16)v;
}

// ---------------------------------------------------------------------------
// Pack an FC weight (N,K) f32 row-major into WMMA B-fragment order, f16:
// Bp[((kt*(N/16)+nt)*32 + lane)*16 + h] = W[nt*16+(lane&15)][kt*32+kmap(lane,h)]
// -> each lane's 16-half fragment is one contiguous, 32B-aligned chunk.
__global__ void pack_wfrag_kernel(const float* __restrict__ w, _Float16* __restrict__ bp,
                                  int K, int N, int total) {
    int id = blockIdx.x * blockDim.x + threadIdx.x;
    if (id >= total) return;
    int h    = id & 15;
    int lane = (id >> 4) & 31;
    int nt   = (id >> 9) % (N >> 4);
    int kt   = id / (512 * (N >> 4));
    int n = nt * 16 + (lane & 15);
    int k = kt * 32 + wmma_kmap(lane, h);
    bp[id] = (_Float16)w[n * K + k];
}

// ---------------------------------------------------------------------------
// WMMA GEMM: C(MxN) = A(MxK,row-major f16) * Bp(fragment-packed f16) + bias.
// One wave = one 16x16 tile; K fully unrolled; two accumulators break the
// WMMA->WMMA RAW chain (5-NOP hazard for f16 WMMA per ISA 7.12.1).
template <int K, int N, bool WF32, bool WF16>
__global__ void gemm_wmma_kernel(const _Float16* __restrict__ A,
                                 const _Float16* __restrict__ Bp,
                                 const float* __restrict__ bias,
                                 float* __restrict__ Cf, _Float16* __restrict__ Ch,
                                 int M) {
    constexpr int KT = K / 32;
    constexpr int NT = N / 16;
    int lane  = threadIdx.x;                               // 0..31 (wave32)
    int tileM = (blockIdx.x * blockDim.y + threadIdx.y) * 16;
    int nt    = blockIdx.y;
    if (tileM >= M) return;                                // wave-uniform
    // A fragment base: row = tileM + (lane&15), half-wave K offset 0 or 8.
    const _Float16* ap = A + (size_t)(tileM + (lane & 15)) * K + ((lane >> 4) << 3);
    const _Float16* bp = Bp + ((size_t)nt * 32 + lane) * 16;
    v8f acc0 = {}, acc1 = {};
    #pragma unroll
    for (int kt = 0; kt < KT; ++kt) {
        v8h alo = *(const v8h*)(ap + kt * 32);
        v8h ahi = *(const v8h*)(ap + kt * 32 + 16);
        v16h a = __builtin_shufflevector(alo, ahi, 0,1,2,3,4,5,6,7,8,9,10,11,12,13,14,15);
        v16h b = *(const v16h*)(bp + (size_t)kt * NT * 512);
        if (kt & 1)
            acc1 = __builtin_amdgcn_wmma_f32_16x16x32_f16(false, a, false, b,
                                                          (short)0, acc1, false, false);
        else
            acc0 = __builtin_amdgcn_wmma_f32_16x16x32_f16(false, a, false, b,
                                                          (short)0, acc0, false, false);
    }
    v8f acc = acc0 + acc1;
    int ncol  = nt * 16 + (lane & 15);
    int mbase = tileM + ((lane >> 4) << 3);                // lanes 16-31 -> M+8
    float bz = bias[ncol];
    #pragma unroll
    for (int r = 0; r < 8; ++r) {
        float v = acc[r] + bz;
        size_t o = (size_t)(mbase + r) * N + ncol;
        if constexpr (WF32) Cf[o] = v;
        if constexpr (WF16) Ch[o] = (_Float16)v;
    }
}

// ---------------------------------------------------------------------------
// FC3: (B,48) @ (6,48)^T + b -> d_out (B,6). Tiny; scalar f32.
__global__ void fc3_kernel(const float* __restrict__ C2, const float* __restrict__ w,
                           const float* __restrict__ bias, float* __restrict__ out, int Bn) {
    int b = blockIdx.x * blockDim.x + threadIdx.x;
    if (b >= Bn) return;
    float xr[48];
    #pragma unroll
    for (int i = 0; i < 48; ++i) xr[i] = C2[(long)b * 48 + i];
    #pragma unroll
    for (int o = 0; o < 6; ++o) {
        float acc = bias[o];
        #pragma unroll
        for (int i = 0; i < 48; ++i) acc += xr[i] * w[o * 48 + i];
        out[(long)b * 6 + o] = acc;
    }
}

// ---------------------------------------------------------------------------
static inline long cdiv(long a, long b) { return (a + b - 1) / b; }

extern "C" void kernel_launch(void* const* d_in, const int* in_sizes, int n_in,
                              void* d_out, int out_size, void* d_ws, size_t ws_size,
                              hipStream_t stream) {
    const int B = 4096;
    const float* x   = (const float*)d_in[0];
    const float* w1  = (const float*)d_in[1];
    const float* w2  = (const float*)d_in[2];
    const float* w3  = (const float*)d_in[3];
    const float* w4  = (const float*)d_in[4];
    const float* g1  = (const float*)d_in[5];
    const float* b1  = (const float*)d_in[6];
    const float* g2  = (const float*)d_in[7];
    const float* b2  = (const float*)d_in[8];
    const float* g3  = (const float*)d_in[9];
    const float* b3  = (const float*)d_in[10];
    const float* g4  = (const float*)d_in[11];
    const float* b4  = (const float*)d_in[12];
    const float* f1w = (const float*)d_in[13];
    const float* f1b = (const float*)d_in[14];
    const float* f2w = (const float*)d_in[15];
    const float* f2b = (const float*)d_in[16];
    const float* f3w = (const float*)d_in[17];
    const float* f3b = (const float*)d_in[18];
    float* out = (float*)d_out;

    // Workspace: region A (88.5 MB) / region B (44.3 MB) ping-pong + small tail.
    char* ws = (char*)d_ws;
    const size_t OFF_B = 88473600;                 // B*12*9*50*4
    const size_t OFF_C = OFF_B + 44236800;         // + B*12*9*25*4
    float*    regA  = (float*)(ws);
    float*    regB  = (float*)(ws + OFF_B);
    float*    stats = (float*)(ws + OFF_C);        // 4 layers x 256 floats
    _Float16* W1t   = (_Float16*)(ws + OFF_C + 4096);
    _Float16* W2t   = (_Float16*)(ws + OFF_C + 4096 + 288 * 96 * 2);

    float* s1 = stats + 0 * 256; float* q1 = s1 + 128;
    float* s2 = stats + 1 * 256; float* q2 = s2 + 128;
    float* s3 = stats + 2 * 256; float* q3 = s3 + 128;
    float* s4 = stats + 3 * 256; float* q4 = s4 + 128;

    zero_stats_kernel<<<4, 256, 0, stream>>>(stats, 4 * 256);

    // ---- Layer 1: lift_conv + pool + relu -> regA (B,12,9,50)
    {
        long total = (long)B * 12 * 9 * 50;
        lift_pool_relu_kernel<<<cdiv(total, 256), 256, 0, stream>>>(x, w1, regA, total);
        stats_kernel<<<4096, 256, 0, stream>>>(regA, s1, q1, 12, 9 * 50, total);
        long tout = (long)B * 12 * 9 * 25;
        bn_pool_kernel<<<cdiv(tout, 256), 256, 0, stream>>>(
            regA, regB, s1, q1, g1, b1, 12, 9, 50, 1.0f / (B * 450.0f), tout);
    }
    // ---- Layer 2: gg_conv w2 + pool + relu -> regA (B,24,7,12)
    {
        long total = (long)B * 24 * 7 * 12;
        ggconv_pool_relu_kernel<3, 5><<<cdiv(total, 256), 256, 0, stream>>>(
            regB, w2, regA, 12, 9, 25, 24, 7, 12, total);
        stats_kernel<<<4096, 256, 0, stream>>>(regA, s2, q2, 24, 7 * 12, total);
        long tout = (long)B * 24 * 7 * 6;
        bn_pool_kernel<<<cdiv(tout, 256), 256, 0, stream>>>(
            regA, regB, s2, q2, g2, b2, 24, 7, 12, 1.0f / (B * 84.0f), tout);
    }
    // ---- Layer 3: gg_conv w3 + pool + relu -> regA (B,48,5,3); BN -> regB
    {
        long total = (long)B * 48 * 5 * 3;
        ggconv_pool_relu_kernel<3, 5><<<cdiv(total, 256), 256, 0, stream>>>(
            regB, w3, regA, 24, 7, 6, 48, 5, 3, total);
        stats_kernel<<<2048, 256, 0, stream>>>(regA, s3, q3, 48, 15, total);
        bn_apply_kernel<<<cdiv(total, 256), 256, 0, stream>>>(
            regA, regB, s3, q3, g3, b3, 48, 15, 1.0f / (B * 15.0f), total);
    }
    // ---- Layer 4: gg_conv w4 + pool + relu -> regA (B,96,3,1); BN+flatten f16 -> regB
    {
        long total = (long)B * 96 * 3 * 1;
        ggconv_pool_relu_kernel<3, 3><<<cdiv(total, 256), 256, 0, stream>>>(
            regB, w4, regA, 48, 5, 3, 96, 3, 1, total);
        stats_kernel<<<1024, 256, 0, stream>>>(regA, s4, q4, 96, 3, total);
        long tout = (long)B * 288;
        bn_flatten_f16_kernel<<<cdiv(tout, 256), 256, 0, stream>>>(
            regA, (_Float16*)regB, s4, q4, g4, b4, 1.0f / (B * 3.0f), tout);
    }
    // ---- FC weights -> f16 fragment-packed
    pack_wfrag_kernel<<<cdiv(288 * 96, 256), 256, 0, stream>>>(f1w, W1t, 288, 96, 288 * 96);
    pack_wfrag_kernel<<<cdiv(96 * 48, 256), 256, 0, stream>>>(f2w, W2t, 96, 48, 96 * 48);

    // ---- FC1: (4096x288)x(288x96)+f1b, WMMA, f16 out -> regA
    {
        dim3 blk(32, 4), grd(B / 64, 96 / 16);
        gemm_wmma_kernel<288, 96, false, true><<<grd, blk, 0, stream>>>(
            (const _Float16*)regB, W1t, f1b, (float*)nullptr, (_Float16*)regA, B);
    }
    // ---- FC2: (4096x96)x(96x48)+f2b, WMMA, f32 out -> regB
    {
        dim3 blk(32, 4), grd(B / 64, 48 / 16);
        gemm_wmma_kernel<96, 48, true, false><<<grd, blk, 0, stream>>>(
            (const _Float16*)regA, W2t, f2b, regB, (_Float16*)nullptr, B);
    }
    // ---- FC3 -> d_out
    fc3_kernel<<<B / 256, 256, 0, stream>>>(regB, f3w, f3b, out, B);
    (void)in_sizes; (void)n_in; (void)out_size; (void)ws_size;
}
